// ObjectRepresentation_6897717477449
// MI455X (gfx1250) — matrix operand order repaired and checked by reading, not compiled, not run on hardware
//
#include <hip/hip_runtime.h>
#include <hip/hip_bf16.h>
#include <math.h>

// ---------------------------------------------------------------------------
// MI455X (gfx1250): ConvLSTM convs as implicit GEMM on v_wmma_f32_16x16x32_bf16.
// Activations staged to LDS as bf16 zero-padded to the 32-wide K granule ->
// A fragments are two ds_load_b128.  Weights prepacked (bf16, K-padded,
// [tap][cout][Kpad]) in workspace -> B fragments are two global_load_b128.
// All fragment-assembly bounds are compile-time (templates) so no per-element
// EXEC predication.  Classifier stays f32 VALU (tiny, protects argmax label).
// ---------------------------------------------------------------------------

typedef __bf16 v16bf __attribute__((ext_vector_type(16)));
typedef __bf16 v8bf  __attribute__((ext_vector_type(8)));
typedef float  v8f   __attribute__((ext_vector_type(8)));

#define HH 512
#define WW 512
#define SLOT (HH * WW * 12)

static __device__ inline __bf16 f2bf(float f) {
    unsigned u = __builtin_bit_cast(unsigned, f);
    unsigned r = u + 0x7FFFu + ((u >> 16) & 1u);   // RNE
    unsigned short h = (unsigned short)(r >> 16);
    return __builtin_bit_cast(__bf16, h);
}

static __device__ inline float hsig(float x) {
    return fminf(fmaxf(0.2f * x + 0.5f, 0.0f), 1.0f);
}

static __device__ inline v16bf cat16(v8bf lo, v8bf hi) {
    return __builtin_shufflevector(lo, hi, 0, 1, 2, 3, 4, 5, 6, 7,
                                            8, 9, 10, 11, 12, 13, 14, 15);
}

// Accumulate one 3x3 conv into acc[0..2] (N tiles 0/16/32).
// LDSARR: bf16 activation tile [3][66][KPAD_], channel-contiguous, zero-padded.
// WPTR:   bf16 packed weights  [tap][48][KPAD_], zero-padded.
#define ACCUM_CONV(LDSARR, WPTR, KPAD_, NCHUNK_)                               \
    for (int tap = 0; tap < 9; ++tap) {                                        \
        const int colb =                                                       \
            ((tap / 3) * 66 + (wv << 4) + mrow + (tap % 3)) * (KPAD_);         \
        _Pragma("unroll")                                                      \
        for (int q = 0; q < (NCHUNK_); ++q) {                                  \
            v8bf alo = *(const v8bf*)&LDSARR[colb + q * 32 + half * 16];       \
            v8bf ahi = *(const v8bf*)&LDSARR[colb + q * 32 + 8 + half * 16];   \
            v16bf afrag = cat16(alo, ahi);                                     \
            _Pragma("unroll")                                                  \
            for (int j = 0; j < 3; ++j) {                                      \
                const __bf16* wb = (WPTR) +                                    \
                    (tap * 48 + (j << 4) + (lane & 15)) * (KPAD_) +            \
                    q * 32 + half * 16;                                        \
                v8bf blo = *(const v8bf*)&wb[0];                               \
                v8bf bhi = *(const v8bf*)&wb[8];                               \
                v16bf bfrag = cat16(blo, bhi);                                 \
                acc[j] = __builtin_amdgcn_wmma_f32_16x16x32_bf16(              \
                    false, afrag, false, bfrag, (short)0, acc[j],              \
                    false, false);                                             \
            }                                                                  \
        }                                                                      \
    }

// ---------------------------------------------------------------------------
// Fused ConvLSTM step.  Block = 128 threads = 4 waves; wave = 16 consecutive
// pixels of one row x 48 out-channels.  Grid = 512 rows x 8 strips.
// ---------------------------------------------------------------------------
template <int CINX, int KPADX>
__global__ __launch_bounds__(128) void convlstm_kernel(
    const float* __restrict__ xa, int xa_ch, int xa_stride, int xa_off,
    const float* __restrict__ xb, int xb_off, int xb_stride,
    const float* __restrict__ hbuf, const float* __restrict__ cbuf,
    const __bf16* __restrict__ wxp_base,   // packed [lab][9][48][KPADX]
    const __bf16* __restrict__ whp_base,   // packed [lab][9][48][32]
    const float* __restrict__ bias,
    const int* __restrict__ label_ptr, int use_label,
    float* __restrict__ hout, int hout_stride, int hout_off,
    float* __restrict__ cout)
{
    __shared__ __attribute__((aligned(16))) __bf16 xs[3 * 66 * KPADX];
    __shared__ __attribute__((aligned(16))) __bf16 hs[3 * 66 * 32];
    __shared__ __attribute__((aligned(16))) float  zs[4 * 16 * 48];

    const int tid = threadIdx.x;
    const int y0 = blockIdx.x >> 3;
    const int x0 = (blockIdx.x & 7) << 6;

    const int lab = use_label ? label_ptr[0] : 0;
    const float*  hb    = hbuf + (size_t)lab * SLOT;
    const float*  cb    = cbuf + (size_t)lab * SLOT;
    const __bf16* wxp   = wxp_base + (size_t)lab * (9 * 48 * KPADX);
    const __bf16* whp   = whp_base + (size_t)lab * (9 * 48 * 32);
    const float*  bp    = bias + lab * 48;
    float*        coutp = cout + (size_t)lab * SLOT;

    // ---- stage x (concat xa ++ xb) as bf16, zero-padded to KPADX ----
    for (int idx = tid; idx < 3 * 66 * KPADX; idx += 128) {
        int ch  = idx % KPADX;
        int pix = idx / KPADX;
        int col = pix % 66;
        int r   = pix / 66;
        int y = y0 + r - 1;
        int x = x0 + col - 1;
        float v = 0.0f;
        if (ch < CINX && y >= 0 && y < HH && x >= 0 && x < WW) {
            int pos = y * WW + x;
            if (ch < xa_ch) v = xa[(size_t)pos * xa_stride + xa_off + ch];
            else            v = xb[(size_t)pos * xb_stride + xb_off + (ch - xa_ch)];
        }
        xs[idx] = f2bf(v);
    }
    // ---- stage h (12 ch) as bf16, zero-padded to 32 ----
    for (int idx = tid; idx < 3 * 66 * 32; idx += 128) {
        int ch  = idx & 31;
        int pix = idx >> 5;
        int col = pix % 66;
        int r   = pix / 66;
        int y = y0 + r - 1;
        int x = x0 + col - 1;
        float v = 0.0f;
        if (ch < 12 && y >= 0 && y < HH && x >= 0 && x < WW)
            v = hb[(size_t)(y * WW + x) * 12 + ch];
        hs[idx] = f2bf(v);
    }
    __syncthreads();

    const int wv   = tid >> 5;
    const int lane = tid & 31;
    const int mrow = lane & 15;
    const int half = lane >> 4;

    v8f acc[3] = {};
    ACCUM_CONV(xs, wxp, KPADX, (KPADX / 32))
    ACCUM_CONV(hs, whp, 32, 1)

    // ---- bias + spill z to LDS (C layout: VGPR r -> M = r + 8*half) ----
    #pragma unroll
    for (int j = 0; j < 3; ++j) {
        const int n = (j << 4) + (lane & 15);
        const float bv = bp[n];
        #pragma unroll
        for (int r = 0; r < 8; ++r)
            zs[wv * 768 + (r + (half << 3)) * 48 + n] = acc[j][r] + bv;
    }
    __syncthreads();

    // ---- gates: 16 pixels x 12 gate-channels per wave = 6 per lane ----
    #pragma unroll
    for (int t = 0; t < 6; ++t) {
        int idx = lane * 6 + t;
        int m  = idx / 12;
        int ch = idx % 12;
        const float* z = &zs[wv * 768 + m * 48];
        float zi = z[ch], zf = z[ch + 12], zc = z[ch + 24], zo = z[ch + 36];
        int pos = y0 * WW + x0 + (wv << 4) + m;
        float cold = cb[(size_t)pos * 12 + ch];
        float cn = hsig(zf) * cold + hsig(zi) * tanhf(zc);
        float hn = hsig(zo) * tanhf(cn);
        hout[(size_t)pos * hout_stride + hout_off + ch] = hn;
        coutp[(size_t)pos * 12 + ch] = cn;
    }
}

// ---------------------------------------------------------------------------
// Weight prepack: f32 [labels][9][cin][48] -> bf16 [labels][9][48][kpad],
// zero-padded in K.  Runs once per launch; output lives in L2 thereafter.
// ---------------------------------------------------------------------------
__global__ __launch_bounds__(256) void pack_weights_kernel(
    const float* __restrict__ w, __bf16* __restrict__ outp,
    int groups /* labels*9 */, int cin, int kpad)
{
    int idx = blockIdx.x * 256 + threadIdx.x;
    int total = groups * 48 * kpad;
    if (idx >= total) return;
    int k = idx % kpad;
    int n = (idx / kpad) % 48;
    int g = idx / (kpad * 48);
    float v = (k < cin) ? w[(g * cin + k) * 48 + n] : 0.0f;
    outp[idx] = f2bf(v);
}

// ---------------------------------------------------------------------------
// Classifier conv (3->32, stride 2, SAME) + ReLU + partial global pool.
// ---------------------------------------------------------------------------
__global__ __launch_bounds__(256) void cls_conv_kernel(
    const float* __restrict__ in, const float* __restrict__ cw,
    const float* __restrict__ cbias, float* __restrict__ pooled, int frame)
{
    __shared__ float pool[32];
    const int tid = threadIdx.x;
    if (tid < 32) pool[tid] = 0.0f;
    __syncthreads();

    const int idx = blockIdx.x * 256 + tid;      // 0 .. 65535
    const int oy = idx >> 8, ox = idx & 255;

    float acc[32];
    #pragma unroll
    for (int co = 0; co < 32; ++co) acc[co] = cbias[co];

    for (int dy = 0; dy < 3; ++dy) {
        int iy = 2 * oy + dy;                    // SAME, stride 2: pad_lo = 0
        if (iy >= HH) continue;
        for (int dx = 0; dx < 3; ++dx) {
            int ix = 2 * ox + dx;
            if (ix >= WW) continue;
            #pragma unroll
            for (int ci = 0; ci < 3; ++ci) {
                float v = in[(size_t)(iy * WW + ix) * 12 + frame * 3 + ci];
                const float* wrow = &cw[((dy * 3 + dx) * 3 + ci) * 32];
                #pragma unroll
                for (int co = 0; co < 32; ++co)
                    acc[co] = fmaf(v, wrow[co], acc[co]);
            }
        }
    }
    #pragma unroll
    for (int co = 0; co < 32; ++co)
        atomicAdd(&pool[co], fmaxf(acc[co], 0.0f));
    __syncthreads();
    if (tid < 32) atomicAdd(&pooled[tid], pool[tid]);
}

// ---------------------------------------------------------------------------
// MLP 32->1024->4 and argmax (softmax is argmax-invariant).
// ---------------------------------------------------------------------------
__global__ __launch_bounds__(1024) void mlp_argmax_kernel(
    const float* __restrict__ pooled,
    const float* __restrict__ d1w, const float* __restrict__ d1b,
    const float* __restrict__ d2w, const float* __restrict__ d2b,
    int* __restrict__ label)
{
    __shared__ float r1[1024];
    __shared__ float pm[32];
    const int t = threadIdx.x;
    if (t < 32) pm[t] = pooled[t] * (1.0f / 65536.0f);
    __syncthreads();
    float a = d1b[t];
    #pragma unroll
    for (int k = 0; k < 32; ++k) a = fmaf(pm[k], d1w[k * 1024 + t], a);
    r1[t] = fmaxf(a, 0.0f);
    __syncthreads();
    if (t == 0) {
        float best = -1e30f; int bi = 0;
        for (int o = 0; o < 4; ++o) {
            float s = d2b[o];
            for (int j = 0; j < 1024; ++j) s = fmaf(r1[j], d2w[j * 4 + o], s);
            if (s > best) { best = s; bi = o; }
        }
        *label = bi;
    }
}

// ---------------------------------------------------------------------------
// Scatter frame's h_new (already in d_out channel block) into h_cls_ws[label].
// ---------------------------------------------------------------------------
__global__ __launch_bounds__(256) void update_h_kernel(
    const float* __restrict__ dout, float* __restrict__ h_ws,
    const int* __restrict__ label_ptr, int frame)
{
    const int lab = label_ptr[0];
    const int idx = blockIdx.x * 256 + threadIdx.x;
    if (idx >= SLOT) return;
    const int pos = idx / 12, ch = idx % 12;
    h_ws[(size_t)lab * SLOT + idx] = dout[(size_t)pos * 48 + frame * 12 + ch];
}

// ---------------------------------------------------------------------------
extern "C" void kernel_launch(void* const* d_in, const int* in_sizes, int n_in,
                              void* d_out, int out_size, void* d_ws, size_t ws_size,
                              hipStream_t stream) {
    (void)in_sizes; (void)n_in; (void)out_size; (void)ws_size;
    const float* inputs     = (const float*)d_in[0];
    const float* h_cls      = (const float*)d_in[1];
    const float* c_cls      = (const float*)d_in[2];
    const float* h_gen      = (const float*)d_in[3];
    const float* c_gen      = (const float*)d_in[4];
    const float* cls_conv_w = (const float*)d_in[5];
    const float* cls_conv_b = (const float*)d_in[6];
    const float* cls_d1_w   = (const float*)d_in[7];
    const float* cls_d1_b   = (const float*)d_in[8];
    const float* cls_d2_w   = (const float*)d_in[9];
    const float* cls_d2_b   = (const float*)d_in[10];
    const float* wx_c       = (const float*)d_in[11];
    const float* wh_c       = (const float*)d_in[12];
    const float* b_c        = (const float*)d_in[13];
    const float* wx_g       = (const float*)d_in[14];
    const float* wh_g       = (const float*)d_in[15];
    const float* b_g        = (const float*)d_in[16];

    char* ws = (char*)d_ws;
    // byte layout (16-aligned sections)
    const size_t OFF_H     = 0;
    const size_t OFF_C     = OFF_H + (size_t)4 * SLOT * 4;
    const size_t OFF_POOL  = OFF_C + (size_t)4 * SLOT * 4;
    const size_t OFF_LABEL = OFF_POOL + 128;
    const size_t OFF_WXC   = OFF_LABEL + 16;                 // 4*9*48*32 bf16
    const size_t OFF_WHC   = OFF_WXC + (size_t)4 * 9 * 48 * 32 * 2;
    const size_t OFF_WXG   = OFF_WHC + (size_t)4 * 9 * 48 * 32 * 2;
    const size_t OFF_WHG   = OFF_WXG + (size_t)9 * 48 * 64 * 2;

    float*  out      = (float*)d_out;
    float*  h_ws     = (float*)(ws + OFF_H);
    float*  c_ws     = (float*)(ws + OFF_C);
    float*  pooled   = (float*)(ws + OFF_POOL);
    int*    label    = (int*)(ws + OFF_LABEL);
    __bf16* wxc_pack = (__bf16*)(ws + OFF_WXC);
    __bf16* whc_pack = (__bf16*)(ws + OFF_WHC);
    __bf16* wxg_pack = (__bf16*)(ws + OFF_WXG);
    __bf16* whg_pack = (__bf16*)(ws + OFF_WHG);

    // Inputs must stay immutable: work on workspace copies of the LSTM state.
    hipMemcpyAsync(h_ws, h_cls, (size_t)4 * SLOT * sizeof(float),
                   hipMemcpyDeviceToDevice, stream);
    hipMemcpyAsync(c_ws, c_cls, (size_t)4 * SLOT * sizeof(float),
                   hipMemcpyDeviceToDevice, stream);

    // Prepack all weights (bf16, K-padded, [tap][cout][Kpad]).
    pack_weights_kernel<<<(4 * 9 * 48 * 32 + 255) / 256, 256, 0, stream>>>(
        wx_c, wxc_pack, 4 * 9, 15, 32);
    pack_weights_kernel<<<(4 * 9 * 48 * 32 + 255) / 256, 256, 0, stream>>>(
        wh_c, whc_pack, 4 * 9, 12, 32);
    pack_weights_kernel<<<(9 * 48 * 64 + 255) / 256, 256, 0, stream>>>(
        wx_g, wxg_pack, 9, 48, 64);
    pack_weights_kernel<<<(9 * 48 * 32 + 255) / 256, 256, 0, stream>>>(
        wh_g, whg_pack, 9, 12, 32);

    for (int f = 0; f < 4; ++f) {
        hipMemsetAsync(pooled, 0, 32 * sizeof(float), stream);
        cls_conv_kernel<<<256, 256, 0, stream>>>(inputs, cls_conv_w, cls_conv_b,
                                                 pooled, f);
        mlp_argmax_kernel<<<1, 1024, 0, stream>>>(pooled, cls_d1_w, cls_d1_b,
                                                  cls_d2_w, cls_d2_b, label);
        // x = concat(h_gen[0] (12ch), frame (3 strided ch of inputs)) -> 15ch
        convlstm_kernel<15, 32><<<4096, 128, 0, stream>>>(
            h_gen, 12, 12, 0,
            inputs, 3 * f, 12,
            h_ws, c_ws,
            wxc_pack, whc_pack, b_c,
            label, 1,
            out, 48, 12 * f,
            c_ws);
        update_h_kernel<<<(SLOT + 255) / 256, 256, 0, stream>>>(out, h_ws,
                                                                label, f);
    }

    // Generator: raw reshape of h_cls -> flat h_ws viewed as (512,512,48) NHWC.
    convlstm_kernel<48, 64><<<4096, 128, 0, stream>>>(
        h_ws, 48, 48, 0,
        nullptr, 0, 48,
        h_gen, c_gen,
        wxg_pack, whg_pack, b_g,
        nullptr, 0,
        out + (size_t)HH * WW * 48, 12, 0,
        out + (size_t)HH * WW * 60);
}
